// LlamaAttentionWithHFRope_24197845745952
// MI455X (gfx1250) — matrix operand order repaired and checked by reading, not compiled, not run on hardware
//
#include <hip/hip_runtime.h>

typedef __bf16 bf16_t;
typedef __attribute__((ext_vector_type(16))) __bf16 v16bf;
typedef __attribute__((ext_vector_type(8)))  __bf16 v8bf;
typedef __attribute__((ext_vector_type(8)))  float  v8f;

#define BATCH     2
#define SEQ       2048
#define DMODEL    4096
#define NHEADS    32
#define NKV       8
#define HEAD_DIM  128
#define DKV       (NKV * HEAD_DIM)      // 1024
#define MROWS     ((size_t)BATCH * SEQ) // 4096

// ---- helpers -------------------------------------------------------------

static __device__ __forceinline__ bf16_t f2bf(float f) {
  // round-to-nearest-even fp32 -> bf16
  unsigned u = __builtin_bit_cast(unsigned, f);
  unsigned r = (u + 0x7FFFu + ((u >> 16) & 1u)) >> 16;
  unsigned short us = (unsigned short)r;
  return __builtin_bit_cast(bf16_t, us);
}

static __device__ __forceinline__ v16bf cat8(v8bf a, v8bf b) {
  return __builtin_shufflevector(a, b, 0,1,2,3,4,5,6,7,8,9,10,11,12,13,14,15);
}

// ---- elementwise fp32 -> bf16 --------------------------------------------

__global__ void cvt_bf16_kernel(const float* __restrict__ X, bf16_t* __restrict__ Xb,
                                long long total) {
  long long idx = (long long)blockIdx.x * blockDim.x + threadIdx.x;
  if (idx >= total) return;
  Xb[idx] = f2bf(X[idx]);
}

// ---- weight transpose + bf16 convert: Wt[n*K + k] = bf16(W[k*N + n]) -----

__global__ void wtrans_kernel(const float* __restrict__ W, bf16_t* __restrict__ Wt,
                              int K, int N) {
  long long idx = (long long)blockIdx.x * blockDim.x + threadIdx.x;
  long long total = (long long)K * N;
  if (idx >= total) return;
  int k = (int)(idx / N);
  int n = (int)(idx % N);
  Wt[(size_t)n * K + k] = f2bf(W[idx]);
}

// ---- GEMM: C[M,N] (f32) = A[M,K] (bf16) * Bt[N,K] (bf16) -----------------
// one wave per 64x64 output tile: 4x4 register blocking, 16 WMMA per K-step

__global__ __launch_bounds__(32)
void gemm_wmma_kernel(const bf16_t* __restrict__ A, const bf16_t* __restrict__ Bt,
                      float* __restrict__ C, int N, int K) {
  const int lane  = threadIdx.x;
  const int m0    = blockIdx.x * 64;
  const int n0    = blockIdx.y * 64;
  const int col   = lane & 15;
  const int baseA = (lane < 16) ? 0 : 16;  // A frag: K = baseA..baseA+15 contiguous
  const int baseB = (lane < 16) ? 0 : 16;  // B frag: K = baseB..baseB+15 contiguous
  const int hi8   = (lane >> 4) * 8;

  const bf16_t* arow[4];
  const bf16_t* brow[4];
#pragma unroll
  for (int t = 0; t < 4; ++t) {
    arow[t] = A  + (size_t)(m0 + t * 16 + col) * K;
    brow[t] = Bt + (size_t)(n0 + t * 16 + col) * K;
  }

  v8f acc[4][4];
#pragma unroll
  for (int mt = 0; mt < 4; ++mt)
#pragma unroll
    for (int nt = 0; nt < 4; ++nt) acc[mt][nt] = (v8f){};

  for (int kk = 0; kk < K; kk += 32) {
    __builtin_prefetch(arow[0] + kk + 512, 0, 0);   // global_prefetch_b8
    __builtin_prefetch(brow[0] + kk + 512, 0, 0);
    v16bf af[4], bf[4];
#pragma unroll
    for (int t = 0; t < 4; ++t) {
      v8bf a0 = *(const v8bf*)(arow[t] + kk + baseA);
      v8bf a1 = *(const v8bf*)(arow[t] + kk + baseA + 8);
      af[t] = cat8(a0, a1);
    }
#pragma unroll
    for (int t = 0; t < 4; ++t) {
      v8bf b0 = *(const v8bf*)(brow[t] + kk + baseB);
      v8bf b1 = *(const v8bf*)(brow[t] + kk + baseB + 8);
      bf[t] = cat8(b0, b1);
    }
#pragma unroll
    for (int mt = 0; mt < 4; ++mt)
#pragma unroll
      for (int nt = 0; nt < 4; ++nt)
        acc[mt][nt] = __builtin_amdgcn_wmma_f32_16x16x32_bf16(
            false, af[mt], false, bf[nt], (short)0, acc[mt][nt], false, false);
  }

#pragma unroll
  for (int mt = 0; mt < 4; ++mt)
#pragma unroll
    for (int nt = 0; nt < 4; ++nt)
#pragma unroll
      for (int r = 0; r < 8; ++r)
        C[(size_t)(m0 + mt * 16 + r + hi8) * N + n0 + nt * 16 + col] = acc[mt][nt][r];
}

// ---- RoPE (HF llama): reads f32 [M, nheads*128], writes bf16 same layout -

__global__ void rope_kernel(const float* __restrict__ Xf, bf16_t* __restrict__ Xb,
                            int nheads) {
  long long idx = (long long)blockIdx.x * blockDim.x + threadIdx.x;
  long long total = (long long)MROWS * nheads * 64;
  if (idx >= total) return;
  int j   = (int)(idx & 63);
  long long t = idx >> 6;
  int h   = (int)(t % nheads);
  int row = (int)(t / nheads);          // b*SEQ + s
  int s   = row % SEQ;
  float invf = __powf(10000.0f, -(float)(2 * j) * (1.0f / (float)HEAD_DIM));
  float ang  = (float)s * invf;
  float c = __cosf(ang), sn = __sinf(ang);
  size_t off = (size_t)row * ((size_t)nheads * HEAD_DIM) + (size_t)h * HEAD_DIM;
  const float* p = Xf + off;
  float x1 = p[j], x2 = p[j + 64];
  bf16_t* q = Xb + off;
  q[j]      = f2bf(x1 * c - x2 * sn);
  q[j + 64] = f2bf(x2 * c + x1 * sn);
}

// ---- V transpose: Vf [M, DKV] f32 -> Vt [(b*NKV+kv)*HEAD_DIM + d][SEQ] bf16

__global__ void vtrans_kernel(const float* __restrict__ Vf, bf16_t* __restrict__ Vt) {
  long long idx = (long long)blockIdx.x * blockDim.x + threadIdx.x;
  long long total = (long long)MROWS * DKV;
  if (idx >= total) return;
  int colv = (int)(idx % DKV);
  int row  = (int)(idx / DKV);
  int b = row / SEQ, s = row % SEQ;
  int kv = colv / HEAD_DIM, d = colv % HEAD_DIM;
  Vt[((size_t)(b * NKV + kv) * HEAD_DIM + d) * SEQ + s] = f2bf(Vf[idx]);
}

// ---- async copy of one K tile (32 keys x 128 dims, bf16) into LDS --------
// 8KB tile = 16 x global_load_async_to_lds_b128 (32 lanes x 16B each)

static __device__ __forceinline__ void ktile_async(const bf16_t* gbase, bf16_t* lbase,
                                                   int lane) {
  const int r2  = lane >> 4;          // 0..1  (2 rows per instruction)
  const int c16 = (lane & 15) * 8;    // 8 bf16 = 16B per lane
#pragma unroll
  for (int i = 0; i < 16; ++i) {
    int row = i * 2 + r2;
    const void* g = (const void*)(gbase + (size_t)row * DKV + c16);
    unsigned l = (unsigned)(size_t)(lbase + row * 128 + c16);
    asm volatile("global_load_async_to_lds_b128 %0, %1, off"
                 :: "v"(l), "v"(g) : "memory");
  }
}

// ---- Flash attention: one wave per (b, h, 16-row q tile) -----------------
// K tiles double-buffered in LDS via async loads; P staged through LDS.

__global__ __launch_bounds__(32)
void attn_kernel(const bf16_t* __restrict__ Qb, const bf16_t* __restrict__ Kb,
                 const bf16_t* __restrict__ Vt, bf16_t* __restrict__ O) {
  __shared__ __align__(16) bf16_t kbuf[2][32 * 128];  // 16KB double buffer
  __shared__ __align__(16) bf16_t pls[16 * 32];       // P tile staging

  const int lane  = threadIdx.x;
  const int qt    = blockIdx.x;          // 0..SEQ/16-1
  const int h     = blockIdx.y;          // 0..31
  const int b     = blockIdx.z;          // 0..1
  const int kvh   = h >> 2;              // n_rep = 4
  const int col   = lane & 15;
  const int baseA = (lane < 16) ? 0 : 16;
  const int baseB = (lane < 16) ? 0 : 16;
  const int hi8   = (lane >> 4) * 8;
  const float scale = 0.08838834764831845f;  // 1/sqrt(128)

  const bf16_t* kgbase = Kb + (size_t)b * SEQ * DKV + (size_t)kvh * HEAD_DIM;

  // Q A-fragments (16 rows x 128 d = 4 chunks of K=32)
  const bf16_t* qrow = Qb + ((size_t)b * SEQ + (size_t)qt * 16 + col) * DMODEL
                          + (size_t)h * HEAD_DIM;
  v16bf qa[4];
#pragma unroll
  for (int c = 0; c < 4; ++c) {
    v8bf q0 = *(const v8bf*)(qrow + c * 32 + baseA);
    v8bf q1 = *(const v8bf*)(qrow + c * 32 + baseA + 8);
    qa[c] = cat8(q0, q1);
  }

  v8f acc[8];
  float rowm[8], rowl[8];
#pragma unroll
  for (int t = 0; t < 8; ++t) acc[t] = (v8f){};
#pragma unroll
  for (int r = 0; r < 8; ++r) { rowm[r] = -3.0e38f; rowl[r] = 0.0f; }

  const int kbmax = (qt * 16 + 15) / 32;   // inclusive; 32 keys per block

  ktile_async(kgbase, kbuf[0], lane);      // prefetch first K tile

  for (int kb = 0; kb <= kbmax; ++kb) {
    if (kb < kbmax) {
      ktile_async(kgbase + (size_t)(kb + 1) * 32 * DKV, kbuf[(kb + 1) & 1], lane);
      asm volatile("s_wait_asynccnt 0x10" ::: "memory");   // current tile resident
    } else {
      asm volatile("s_wait_asynccnt 0x0" ::: "memory");
    }
    const bf16_t* kt = kbuf[kb & 1];

    // ---- S = Q K^T for 32 keys (two 16-col halves) ----
    float sv[2][8];
#pragma unroll
    for (int half = 0; half < 2; ++half) {
      int key = kb * 32 + half * 16 + col;
      const bf16_t* krow = kt + (half * 16 + col) * 128;
      v8f sacc = {};
#pragma unroll
      for (int c = 0; c < 4; ++c) {
        v8bf k0 = *(const v8bf*)(krow + c * 32 + baseB);
        v8bf k1 = *(const v8bf*)(krow + c * 32 + baseB + 8);
        sacc = __builtin_amdgcn_wmma_f32_16x16x32_bf16(false, qa[c], false, cat8(k0, k1),
                                                       (short)0, sacc, false, false);
      }
#pragma unroll
      for (int r = 0; r < 8; ++r) {
        int qg = qt * 16 + r + hi8;
        sv[half][r] = (key <= qg) ? sacc[r] * scale : -3.0e38f;
      }
    }
    // ---- online softmax (row stats across 16 lanes per half) ----
    float co[8];
#pragma unroll
    for (int r = 0; r < 8; ++r) {
      float mloc = fmaxf(sv[0][r], sv[1][r]);
#pragma unroll
      for (int off = 1; off < 16; off <<= 1)
        mloc = fmaxf(mloc, __shfl_xor(mloc, off, 16));
      float mnew = fmaxf(rowm[r], mloc);
      co[r] = __expf(rowm[r] - mnew);
      rowm[r] = mnew;
    }
#pragma unroll
    for (int r = 0; r < 8; ++r) {
      float p0 = __expf(sv[0][r] - rowm[r]);
      float p1 = __expf(sv[1][r] - rowm[r]);
      pls[(r + hi8) * 32 + col]      = f2bf(p0);
      pls[(r + hi8) * 32 + 16 + col] = f2bf(p1);
      float sloc = p0 + p1;
#pragma unroll
      for (int off = 1; off < 16; off <<= 1)
        sloc += __shfl_xor(sloc, off, 16);
      rowl[r] = rowl[r] * co[r] + sloc;
    }
#pragma unroll
    for (int t = 0; t < 8; ++t)
#pragma unroll
      for (int r = 0; r < 8; ++r) acc[t][r] *= co[r];

    // ---- P A-fragment from LDS (same-wave DS ops are in order) ----
    v8bf p0 = *(const v8bf*)(pls + col * 32 + baseA);
    v8bf p1 = *(const v8bf*)(pls + col * 32 + baseA + 8);
    v16bf pa = cat8(p0, p1);

    // ---- O += P * V (V^T rows contiguous over keys) ----
#pragma unroll
    for (int t = 0; t < 8; ++t) {
      const bf16_t* vrow = Vt + ((size_t)(b * NKV + kvh) * HEAD_DIM + t * 16 + col) * SEQ
                              + (size_t)kb * 32;
      v8bf v0 = *(const v8bf*)(vrow + baseB);
      v8bf v1 = *(const v8bf*)(vrow + baseB + 8);
      acc[t] = __builtin_amdgcn_wmma_f32_16x16x32_bf16(false, pa, false, cat8(v0, v1),
                                                       (short)0, acc[t], false, false);
    }
  }

  // ---- epilogue: normalize and store bf16 [M, DMODEL] ----
  bf16_t* orow = O + ((size_t)b * SEQ + (size_t)qt * 16) * DMODEL + (size_t)h * HEAD_DIM;
#pragma unroll
  for (int t = 0; t < 8; ++t)
#pragma unroll
    for (int r = 0; r < 8; ++r)
      orow[(size_t)(r + hi8) * DMODEL + t * 16 + col] = f2bf(acc[t][r] / rowl[r]);
}

// ---- host launch ---------------------------------------------------------

extern "C" void kernel_launch(void* const* d_in, const int* in_sizes, int n_in,
                              void* d_out, int out_size, void* d_ws, size_t ws_size,
                              hipStream_t stream) {
  (void)in_sizes; (void)n_in; (void)out_size; (void)ws_size;
  const float* x  = (const float*)d_in[0];
  const float* Wq = (const float*)d_in[1];
  const float* Wk = (const float*)d_in[2];
  const float* Wv = (const float*)d_in[3];
  const float* Wo = (const float*)d_in[4];
  float* out = (float*)d_out;

  char* ws = (char*)d_ws;
  size_t off = 0;
  auto alloc = [&](size_t bytes) -> void* {
    void* p = ws + off;
    off = (off + bytes + 255) & ~(size_t)255;
    return p;
  };
  float*  Qf    = (float*) alloc(MROWS * DMODEL * 4);   // reused for attn-out bf16
  float*  Kf    = (float*) alloc(MROWS * DKV * 4);
  float*  Vf    = (float*) alloc(MROWS * DKV * 4);
  bf16_t* xbf   = (bf16_t*)alloc(MROWS * DMODEL * 2);
  bf16_t* Qbf   = (bf16_t*)alloc(MROWS * DMODEL * 2);
  bf16_t* Kbf   = (bf16_t*)alloc(MROWS * DKV * 2);
  bf16_t* Vtb   = (bf16_t*)alloc(MROWS * DKV * 2);
  bf16_t* WqT   = (bf16_t*)alloc((size_t)DMODEL * DMODEL * 2);
  bf16_t* WkT   = (bf16_t*)alloc((size_t)DMODEL * DKV * 2);
  bf16_t* WvT   = (bf16_t*)alloc((size_t)DMODEL * DKV * 2);
  bf16_t* WoT   = (bf16_t*)alloc((size_t)DMODEL * DMODEL * 2);
  bf16_t* attnb = (bf16_t*)Qf;   // Qf is dead after RoPE; 32MB fits in 64MB slot

  auto cdiv = [](long long a, long long b) -> unsigned { return (unsigned)((a + b - 1) / b); };

  // activations + weights -> bf16 (weights transposed to [N,K])
  cvt_bf16_kernel<<<cdiv((long long)MROWS * DMODEL, 256), 256, 0, stream>>>(x, xbf,
                                                       (long long)MROWS * DMODEL);
  wtrans_kernel<<<cdiv((long long)DMODEL * DMODEL, 256), 256, 0, stream>>>(Wq, WqT, DMODEL, DMODEL);
  wtrans_kernel<<<cdiv((long long)DMODEL * DKV,    256), 256, 0, stream>>>(Wk, WkT, DMODEL, DKV);
  wtrans_kernel<<<cdiv((long long)DMODEL * DKV,    256), 256, 0, stream>>>(Wv, WvT, DMODEL, DKV);
  wtrans_kernel<<<cdiv((long long)DMODEL * DMODEL, 256), 256, 0, stream>>>(Wo, WoT, DMODEL, DMODEL);

  // QKV projections (WMMA, 64x64 per wave)
  gemm_wmma_kernel<<<dim3(MROWS / 64, DMODEL / 64), 32, 0, stream>>>(xbf, WqT, Qf, DMODEL, DMODEL);
  gemm_wmma_kernel<<<dim3(MROWS / 64, DKV    / 64), 32, 0, stream>>>(xbf, WkT, Kf, DKV,    DMODEL);
  gemm_wmma_kernel<<<dim3(MROWS / 64, DKV    / 64), 32, 0, stream>>>(xbf, WvT, Vf, DKV,    DMODEL);

  // RoPE + layout conversions
  rope_kernel<<<cdiv((long long)MROWS * NHEADS * 64, 256), 256, 0, stream>>>(Qf, Qbf, NHEADS);
  rope_kernel<<<cdiv((long long)MROWS * NKV    * 64, 256), 256, 0, stream>>>(Kf, Kbf, NKV);
  vtrans_kernel<<<cdiv((long long)MROWS * DKV, 256), 256, 0, stream>>>(Vf, Vtb);

  // flash attention (WMMA + async K-tile staging); bf16 out into Qf slot
  attn_kernel<<<dim3(SEQ / 16, NHEADS, BATCH), 32, 0, stream>>>(Qbf, Kbf, Vtb, attnb);

  // output projection (WMMA)
  gemm_wmma_kernel<<<dim3(MROWS / 64, DMODEL / 64), 32, 0, stream>>>(attnb, WoT, out, DMODEL, DMODEL);
}
